// ChebyKANLayer_72464688218723
// MI455X (gfx1250) — matrix operand order repaired and checked by reading, not compiled, not run on hardware
//
#include <hip/hip_runtime.h>
#include <hip/hip_bf16.h>

typedef __attribute__((ext_vector_type(16))) __bf16 v16bf;
typedef __attribute__((ext_vector_type(8)))  float  v8f;

constexpr int BATCH = 16384;
constexpr int IN    = 1024;
constexpr int OUT   = 1024;
constexpr int DEG1  = 9;      // DEG + 1
constexpr int TM    = 128;    // M tile per workgroup
constexpr int TN    = 128;    // N tile per workgroup
constexpr int KS    = 32;     // K slab == WMMA K
constexpr int LDA   = 40;     // LDS row stride (bf16), 80B: 16B aligned, bank-spread

// CDNA5 has a hardware V_TANH_F32 (listed as a TRANS op in the ISA).
#if __has_builtin(__builtin_amdgcn_tanhf)
__device__ __forceinline__ float fast_tanh(float v) { return __builtin_amdgcn_tanhf(v); }
#else
__device__ __forceinline__ float fast_tanh(float v) { return tanhf(v); }
#endif

// ---------------------------------------------------------------------------
// Kernel 1: coeffs (I,O,9) f32  ->  Wt[d][o][i] bf16  (18.9 MB, L2-resident)
// ---------------------------------------------------------------------------
__global__ __launch_bounds__(256) void cheby_transpose(const float* __restrict__ C,
                                                       __bf16* __restrict__ Wt) {
  const int idx = blockIdx.x * 256 + threadIdx.x;   // over I*O
  const int i = idx >> 10;                          // / OUT
  const int o = idx & (OUT - 1);
  const float* src = C + ((size_t)i * OUT + o) * DEG1;
#pragma unroll
  for (int d = 0; d < DEG1; ++d)
    Wt[((size_t)d * OUT + o) * IN + i] = (__bf16)src[d];
}

// ---------------------------------------------------------------------------
// Kernel 2: fused Chebyshev-basis + bf16 WMMA GEMM
//   y[b,o] = sum_d sum_i T_d(tanh(x[b,i])) * Wt[d][o][i]
// ---------------------------------------------------------------------------
__global__ __launch_bounds__(256) void cheby_wmma_gemm(const float* __restrict__ x,
                                                       const __bf16* __restrict__ Wt,
                                                       float* __restrict__ y) {
  // Double-buffered 128x32 bf16 basis tile: one barrier per degree step.
  __shared__ __align__(16) __bf16 As[2][TM * LDA];  // 20 KB

  const int tid  = threadIdx.x;
  const int lane = tid & 31;
  const int wave = tid >> 5;
  const int wm   = (wave >> 1) * 32;                // 4 waves along M
  const int wn   = (wave & 1) * 64;                 // 2 waves along N
  const int hl   = lane >> 4;
  const int ln   = lane & 15;
  const int m_base = blockIdx.x * TM;
  const int n_base = blockIdx.y * TN;

  v8f acc[2][4] = {};                               // 2(M) x 4(N) 16x16 f32 tiles
  float tt[16], Tp[16], Tc[16], cv[16];             // column-pair recurrence state

  // Each thread owns 8 column-pairs: pair p -> row = (tid+256p)>>4,
  // cols = (tid&15)*2, (tid&15)*2+1  ->  float2 loads, packed b32 LDS stores.
  auto store_basis = [&](__bf16* buf, const float* v) {
#pragma unroll
    for (int p = 0; p < 8; ++p) {
      const int ep  = tid + 256 * p;
      const int row = ep >> 4;
      const int col = (ep & 15) * 2;
      union { __bf16 b[2]; unsigned u; } pk;
      pk.b[0] = (__bf16)v[2 * p];
      pk.b[1] = (__bf16)v[2 * p + 1];
      *(unsigned*)&buf[row * LDA + col] = pk.u;     // ds_store_b32 (v_cvt_pk_bf16_f32)
    }
  };

  auto mma_step = [&](int d, int k0, const __bf16* buf) {
    // A frags: lane row = lane&15; V0-3 = K hl*8+0..7, V4-7 = K 16+hl*8+0..7.
    v16bf afrag[2];
#pragma unroll
    for (int mt = 0; mt < 2; ++mt) {
      const __bf16* ab = &buf[(wm + mt * 16 + ln) * LDA + hl * 8];
      ((float4*)&afrag[mt])[0] = *(const float4*)(ab);
      ((float4*)&afrag[mt])[1] = *(const float4*)(ab + 16);
    }
    // B frags from L2-resident Wt: lane col = lane&15; V0-3 = K hl*8+0..7, V4-7 = K 16+hl*8+0..7.
#pragma unroll
    for (int nt = 0; nt < 4; ++nt) {
      const int col = n_base + wn + nt * 16 + ln;
      const __bf16* bb = Wt + ((size_t)d * OUT + col) * IN + k0 + hl * 8;
      v16bf bfrag;
      ((float4*)&bfrag)[0] = *(const float4*)(bb);
      ((float4*)&bfrag)[1] = *(const float4*)(bb + 16);
#pragma unroll
      for (int mt = 0; mt < 2; ++mt)
        acc[mt][nt] = __builtin_amdgcn_wmma_f32_16x16x32_bf16(
            false, afrag[mt], false, bfrag, (short)0, acc[mt][nt], false, false);
    }
  };

  int pb = 0;                                       // LDS buffer parity (DEG1 odd ->
                                                    // alternates cleanly across slabs)
  for (int k0 = 0; k0 < IN; k0 += KS) {
    // Prefetch next slab's x tile into cache (global_prefetch, no counter).
    if (k0 + KS < IN) {
      const float* pf = &x[(size_t)(m_base + (tid >> 1)) * IN + (k0 + KS) + (tid & 1) * 16];
      __builtin_prefetch(pf, 0, 0);
    }
    // Load 128x32 x tile as float2 pairs, tanh once per slab.
#pragma unroll
    for (int p = 0; p < 8; ++p) {
      const int ep  = tid + 256 * p;
      const int row = ep >> 4;
      const int col = (ep & 15) * 2;
      const float2 xv = *(const float2*)&x[(size_t)(m_base + row) * IN + k0 + col];
      tt[2 * p]     = fast_tanh(xv.x);
      tt[2 * p + 1] = fast_tanh(xv.y);
    }
    // d = 0 : T_0 = 1
#pragma unroll
    for (int j = 0; j < 16; ++j) cv[j] = 1.0f;
    store_basis(As[pb], cv);
    __syncthreads();
    mma_step(0, k0, As[pb]);
    pb ^= 1;
    // d = 1 : T_1 = t
#pragma unroll
    for (int j = 0; j < 16; ++j) { Tp[j] = 1.0f; Tc[j] = tt[j]; }
    store_basis(As[pb], tt);
    __syncthreads();
    mma_step(1, k0, As[pb]);
    pb ^= 1;
    // d >= 2 : T_d = 2 t T_{d-1} - T_{d-2}
    for (int d = 2; d < DEG1; ++d) {
#pragma unroll
      for (int j = 0; j < 16; ++j) {
        cv[j] = 2.0f * tt[j] * Tc[j] - Tp[j];
        Tp[j] = Tc[j];
        Tc[j] = cv[j];
      }
      store_basis(As[pb], cv);
      __syncthreads();
      mma_step(d, k0, As[pb]);
      pb ^= 1;
    }
  }

  // Epilogue: C/D layout -> VGPR rv holds row hl*8+rv, col = lane&15.
#pragma unroll
  for (int mt = 0; mt < 2; ++mt)
#pragma unroll
    for (int nt = 0; nt < 4; ++nt)
#pragma unroll
      for (int rv = 0; rv < 8; ++rv) {
        const int row = m_base + wm + mt * 16 + hl * 8 + rv;
        const int col = n_base + wn + nt * 16 + ln;
        y[(size_t)row * OUT + col] = acc[mt][nt][rv];
      }
}

// ---------------------------------------------------------------------------
extern "C" void kernel_launch(void* const* d_in, const int* in_sizes, int n_in,
                              void* d_out, int out_size, void* d_ws, size_t ws_size,
                              hipStream_t stream) {
  (void)in_sizes; (void)n_in; (void)out_size; (void)ws_size;
  const float* x      = (const float*)d_in[0];
  const float* coeffs = (const float*)d_in[1];   // (I, O, 9)
  float* y            = (float*)d_out;           // (B, O)
  __bf16* Wt          = (__bf16*)d_ws;           // needs 9*1024*1024*2 = 18.9 MB

  cheby_transpose<<<dim3((IN * OUT) / 256), dim3(256), 0, stream>>>(coeffs, Wt);
  cheby_wmma_gemm<<<dim3(BATCH / TM, OUT / TN), dim3(256), 0, stream>>>(x, Wt, y);
}
